// GRUModel_65025804861878
// MI455X (gfx1250) — matrix operand order repaired and checked by reading, not compiled, not run on hardware
//
#include <hip/hip_runtime.h>

// ---- problem constants (from reference) ----
constexpr int kN = 10000;
constexpr int kS = 20;
constexpr int kH = 512;
constexpr int kL = 2;
constexpr int kE = 100000;

typedef __attribute__((ext_vector_type(16))) __bf16 v16bf;
typedef __attribute__((ext_vector_type(8)))  float  v8f;

union FragBF {
    v16bf bf;
    uint4 q[2];
};

__device__ __forceinline__ unsigned short f32_to_bf16(float f) {
    unsigned int u = __float_as_uint(f);
    unsigned int r = (u + 0x7FFFu + ((u >> 16) & 1u)) >> 16;   // round-to-nearest-even
    return (unsigned short)r;
}

// fast sigmoid: v_exp_f32 + v_rcp_f32 (avoids precise-division expansion)
__device__ __forceinline__ float fast_sigmoid(float x) {
    return __builtin_amdgcn_rcpf(1.0f + __expf(-x));
}

// ---------------------------------------------------------------------------
// Kernel 1: fp32 -> bf16 convert (weights)
// ---------------------------------------------------------------------------
__global__ void cvt_bf16_kernel(const float* __restrict__ src,
                                unsigned short* __restrict__ dst, int count) {
    int i = blockIdx.x * 256 + threadIdx.x;
    if (i < count) dst[i] = f32_to_bf16(src[i]);
}

// ---------------------------------------------------------------------------
// Kernel 2: gather x = emb[walk_idx] and convert to bf16, layout [S][N][H]
// ---------------------------------------------------------------------------
__global__ void gather_x_kernel(const float* __restrict__ emb,
                                const int* __restrict__ walk_idx,
                                unsigned short* __restrict__ xbf) {
    long long t = (long long)blockIdx.x * 256 + threadIdx.x;       // over S*N*H/4
    const long long total = (long long)kS * kN * (kH / 4);
    if (t >= total) return;
    int c4  = (int)(t % (kH / 4));
    long long row = t / (kH / 4);          // [s][n]
    int s = (int)(row / kN);
    int n = (int)(row % kN);
    int v = walk_idx[(long long)n * kS + s];
    const float4 f = ((const float4*)(emb + (long long)v * kH))[c4];
    unsigned short o0 = f32_to_bf16(f.x), o1 = f32_to_bf16(f.y);
    unsigned short o2 = f32_to_bf16(f.z), o3 = f32_to_bf16(f.w);
    unsigned int p0 = (unsigned int)o0 | ((unsigned int)o1 << 16);
    unsigned int p1 = (unsigned int)o2 | ((unsigned int)o3 << 16);
    uint2* dst = (uint2*)(xbf + (((long long)s * kN + n) * kH) + c4 * 4);
    *dst = make_uint2(p0, p1);
}

// ---------------------------------------------------------------------------
// Kernel 3: fused WMMA GRU with async global->LDS double buffering.
//   grid  = (N/16, 512/FS)   block = 256 (8 waves)
//   Weight slice (2 gates x 128 x 512 bf16) LDS-resident per layer;
//   x step-tiles streamed via GLOBAL_LOAD_ASYNC_TO_LDS_B128 (ASYNCcnt),
//   overlapped with the WMMA loop of the current step. hn stays in registers
//   across both layers.
// ---------------------------------------------------------------------------
constexpr int FS   = 128;         // features per workgroup
constexpr int HPAD = kH + 8;      // +16B row pad -> conflict-free b128 LDS reads

__global__ __launch_bounds__(256) void gru_kernel(
        const unsigned short* __restrict__ xbf,   // [S][N][H] bf16
        const unsigned short* __restrict__ wir,   // [L][H][H] bf16
        const unsigned short* __restrict__ wiz,   // [L][H][H] bf16
        const float* __restrict__ b_ir,           // [L][H]
        const float* __restrict__ b_iz,           // [L][H]
        float* __restrict__ hn_out)               // [N][H]
{
    __shared__ unsigned short wsl[2][FS][HPAD];   // ~266 KB
    __shared__ unsigned short xs[2][16][HPAD];    // ~33 KB (double buffered)

    const int tid  = threadIdx.x;
    const int wave = tid >> 5;
    const int lane = tid & 31;
    const int m    = lane & 15;
    const int half = lane >> 4;
    const int n0   = blockIdx.x * 16;
    const int f0   = blockIdx.y * FS;
    const int featLocal  = wave * 16 + m;         // 0..127
    const int featGlobal = f0 + featLocal;

    // issue async b128 loads of one 16x512 bf16 step-tile into xs[buf]
    auto issue_async = [&](int s, int buf) {
        const unsigned short* base = xbf + ((long long)s * kN + n0) * kH;
        #pragma unroll
        for (int j = 0; j < 4; ++j) {
            int i = tid + j * 256;                 // 1024 uint4 chunks total
            int r = i >> 6;                        // / (kH/8)
            int c = (i & 63) * 8;
            unsigned lds  = (unsigned)(uintptr_t)&xs[buf][r][c];  // LDS addr = low 32 bits
            unsigned goff = (unsigned)(i * 16);
            asm volatile("global_load_async_to_lds_b128 %0, %1, %2"
                         :: "v"(lds), "v"(goff), "s"(base) : "memory");
        }
    };

    v8f hn = {};
    int buf = 0;

    for (int l = 0; l < kL; ++l) {
        __syncthreads();   // previous layer fully done before overwriting wsl
        // ---- stage weight slice for this layer into LDS (both gates) ----
        {
            const uint4* s0 = (const uint4*)(wir + ((long long)l * kH + f0) * kH);
            const uint4* s1 = (const uint4*)(wiz + ((long long)l * kH + f0) * kH);
            const int nq = FS * kH / 8;           // uint4 per gate
            for (int i = tid; i < nq; i += 256) {
                int r = i >> 6;
                int c = (i & 63) * 8;
                *(uint4*)&wsl[0][r][c] = s0[i];
                *(uint4*)&wsl[1][r][c] = s1[i];
            }
        }
        issue_async(0, buf);                       // first step tile of this layer
        const float bir = b_ir[l * kH + featGlobal];
        const float biz = b_iz[l * kH + featGlobal];
        asm volatile("s_wait_asynccnt 0x0" ::: "memory");
        __syncthreads();

        for (int s = 0; s < kS; ++s) {
            if (s + 1 < kS)
                issue_async(s + 1, buf ^ 1);       // overlap next tile with compute

            v8f hacc = {};
            v8f zacc = {};
            #pragma unroll 4
            for (int kc = 0; kc < kH / 32; ++kc) {
                FragBF a, br, bz;
                // A: 16x32 bf16 fragment (K contiguous: K=half*16..half*16+15)
                const unsigned short* arow = &xs[buf][m][kc * 32 + half * 16];
                a.q[0] = *(const uint4*)(arow);
                a.q[1] = *(const uint4*)(arow + 8);
                // B: 32x16 bf16 fragments (lane = out feature, K contiguous)
                const unsigned short* r0 = &wsl[0][featLocal][kc * 32 + half * 16];
                const unsigned short* r1 = &wsl[1][featLocal][kc * 32 + half * 16];
                br.q[0] = *(const uint4*)(r0);
                br.q[1] = *(const uint4*)(r0 + 8);
                bz.q[0] = *(const uint4*)(r1);
                bz.q[1] = *(const uint4*)(r1 + 8);
                hacc = __builtin_amdgcn_wmma_f32_16x16x32_bf16(
                           false, a.bf, false, br.bf, (short)0, hacc, false, false);
                zacc = __builtin_amdgcn_wmma_f32_16x16x32_bf16(
                           false, a.bf, false, bz.bf, (short)0, zacc, false, false);
            }
            // ---- fused GRU scan step: hn = (1-z)*hn + z*h ----
            #pragma unroll
            for (int i = 0; i < 8; ++i) {
                float h = hacc[i] + bir;
                float z = fast_sigmoid(zacc[i] + biz);
                hn[i] = (1.0f - z) * hn[i] + z * h;
            }
            // wave's own async loads (for s+1) landed in LDS, then block-wide sync
            asm volatile("s_wait_asynccnt 0x0" ::: "memory");
            __syncthreads();
            buf ^= 1;
        }
    }

    // write hn tile: VGPR i -> node row n0 + half*8 + i, col featGlobal
    #pragma unroll
    for (int i = 0; i < 8; ++i) {
        int row = n0 + half * 8 + i;
        hn_out[(long long)row * kH + featGlobal] = hn[i];
    }
}

// ---------------------------------------------------------------------------
// Kernel 4: score[n] = sigmoid(hn[n] . lin_w + lin_b); also hn -> bf16
// ---------------------------------------------------------------------------
__global__ void score_kernel(const float* __restrict__ hn,
                             const float* __restrict__ lin_w,
                             const float* __restrict__ lin_b,
                             float* __restrict__ score,
                             unsigned short* __restrict__ hnbf) {
    int n = blockIdx.x * 8 + (threadIdx.x >> 5);
    int lane = threadIdx.x & 31;
    if (n >= kN) return;
    const float* row = hn + (long long)n * kH;
    float sum = 0.0f;
    for (int i = lane; i < kH; i += 32) {
        float v = row[i];
        sum += v * lin_w[i];
        hnbf[(long long)n * kH + i] = f32_to_bf16(v);
    }
    for (int off = 16; off > 0; off >>= 1)
        sum += __shfl_xor(sum, off, 32);
    if (lane == 0)
        score[n] = fast_sigmoid(sum + lin_b[0]);
}

// ---------------------------------------------------------------------------
// Kernel 5: pred[e] = score[src[e]] * score[dst[e]]
// ---------------------------------------------------------------------------
__global__ void pred_kernel(const int* __restrict__ eli,
                            const float* __restrict__ score,
                            float* __restrict__ pred) {
    int e = blockIdx.x * 256 + threadIdx.x;
    if (e < kE) pred[e] = score[eli[e]] * score[eli[kE + e]];
}

// ---------------------------------------------------------------------------
// Kernel 6: out2 = hn @ skip_w^T + skip_b via WMMA (bf16 in, f32 acc)
// ---------------------------------------------------------------------------
__global__ __launch_bounds__(256) void out2_kernel(
        const unsigned short* __restrict__ hnbf,   // [N][H]
        const unsigned short* __restrict__ skw,    // [H][H]
        const float* __restrict__ skip_b,
        float* __restrict__ out2)                  // [N][H]
{
    const int wave = threadIdx.x >> 5;
    const int lane = threadIdx.x & 31;
    const int m    = lane & 15;
    const int half = lane >> 4;
    const int n0   = blockIdx.x * 16;
    const int featGlobal = blockIdx.y * 128 + wave * 16 + m;

    const unsigned short* arow = hnbf + (long long)(n0 + m) * kH;
    const unsigned short* brow = skw + (long long)featGlobal * kH;

    v8f acc = {};
    #pragma unroll 4
    for (int kc = 0; kc < kH / 32; ++kc) {
        FragBF a, b;
        const unsigned short* pa = arow + kc * 32 + half * 16;
        a.q[0] = *(const uint4*)(pa);
        a.q[1] = *(const uint4*)(pa + 8);
        const unsigned short* pb = brow + kc * 32 + half * 16;
        b.q[0] = *(const uint4*)(pb);
        b.q[1] = *(const uint4*)(pb + 8);
        acc = __builtin_amdgcn_wmma_f32_16x16x32_bf16(
                  false, a.bf, false, b.bf, (short)0, acc, false, false);
    }
    float bias = skip_b[featGlobal];
    #pragma unroll
    for (int i = 0; i < 8; ++i) {
        int row = n0 + half * 8 + i;
        out2[(long long)row * kH + featGlobal] = acc[i] + bias;
    }
}

// ---------------------------------------------------------------------------
extern "C" void kernel_launch(void* const* d_in, const int* in_sizes, int n_in,
                              void* d_out, int out_size, void* d_ws, size_t ws_size,
                              hipStream_t stream) {
    const float* emb      = (const float*)d_in[0];
    const float* W_ir     = (const float*)d_in[1];
    const float* b_ir     = (const float*)d_in[2];
    const float* W_iz     = (const float*)d_in[3];
    const float* b_iz     = (const float*)d_in[4];
    const float* lin_w    = (const float*)d_in[5];
    const float* lin_b    = (const float*)d_in[6];
    const float* skip_w   = (const float*)d_in[7];
    const float* skip_b   = (const float*)d_in[8];
    const int*   walk_idx = (const int*)d_in[9];
    const int*   eli      = (const int*)d_in[10];

    // ---- workspace layout ----
    unsigned char* base = (unsigned char*)d_ws;
    size_t off = 0;
    auto alloc = [&](size_t bytes) -> void* {
        off = (off + 255) & ~(size_t)255;
        void* p = base + off;
        off += bytes;
        return p;
    };
    unsigned short* xbf   = (unsigned short*)alloc((size_t)kS * kN * kH * 2);  // 204.8 MB
    unsigned short* wirbf = (unsigned short*)alloc((size_t)kL * kH * kH * 2);
    unsigned short* wizbf = (unsigned short*)alloc((size_t)kL * kH * kH * 2);
    unsigned short* skwbf = (unsigned short*)alloc((size_t)kH * kH * 2);
    float*          hn    = (float*)alloc((size_t)kN * kH * 4);
    unsigned short* hnbf  = (unsigned short*)alloc((size_t)kN * kH * 2);
    float*          score = (float*)alloc((size_t)kN * 4);
    (void)ws_size; (void)in_sizes; (void)n_in; (void)out_size;

    float* pred = (float*)d_out;          // [E]
    float* out2 = (float*)d_out + kE;     // [N][H]

    // 1) convert weights to bf16
    {
        int c1 = kL * kH * kH;
        cvt_bf16_kernel<<<(c1 + 255) / 256, 256, 0, stream>>>(W_ir, wirbf, c1);
        cvt_bf16_kernel<<<(c1 + 255) / 256, 256, 0, stream>>>(W_iz, wizbf, c1);
        int c2 = kH * kH;
        cvt_bf16_kernel<<<(c2 + 255) / 256, 256, 0, stream>>>(skip_w, skwbf, c2);
    }
    // 2) gather + convert x
    {
        long long total = (long long)kS * kN * (kH / 4);
        gather_x_kernel<<<(unsigned)((total + 255) / 256), 256, 0, stream>>>(emb, walk_idx, xbf);
    }
    // 3) fused WMMA GRU (async double-buffered x tiles)
    gru_kernel<<<dim3(kN / 16, kH / FS), 256, 0, stream>>>(xbf, wirbf, wizbf, b_ir, b_iz, hn);
    // 4) score + hn->bf16
    score_kernel<<<(kN + 7) / 8, 256, 0, stream>>>(hn, lin_w, lin_b, score, hnbf);
    // 5) edge predictions
    pred_kernel<<<(kE + 255) / 256, 256, 0, stream>>>(eli, score, pred);
    // 6) skip projection via WMMA
    out2_kernel<<<dim3(kN / 16, kH / 128), 256, 0, stream>>>(hnbf, skwbf, skip_b, out2);
}
